// MatrixOdeTrainableModelLeastSquares_4861902979218
// MI455X (gfx1250) — matrix operand order repaired and checked by reading, not compiled, not run on hardware
//
#include <hip/hip_runtime.h>

// ---------------------------------------------------------------------------
// Types for CDNA5 WMMA fragments
// ---------------------------------------------------------------------------
typedef __attribute__((ext_vector_type(2)))  float          v2f;
typedef __attribute__((ext_vector_type(4)))  float          v4f;
typedef __attribute__((ext_vector_type(8)))  float          v8f;
typedef __attribute__((ext_vector_type(16))) __bf16         v16bf;
typedef __attribute__((ext_vector_type(16))) unsigned short v16u;

static __device__ __forceinline__ unsigned short f32_to_bf16_rne(float f) {
  union { float f; unsigned u; } v; v.f = f;
  unsigned u = v.u;
  unsigned r = u + 0x7fffu + ((u >> 16) & 1u);   // round-to-nearest-even
  return (unsigned short)(r >> 16);
}

static __device__ __forceinline__ v16bf pack_bf16(const v16u& bits) {
  union { v16u u; v16bf bf; } cvt; cvt.u = bits; return cvt.bf;
}

// Low 32 bits of a generic pointer to a __shared__ object == LDS byte offset
// (flat shared aperture keeps the group-segment offset in addr[31:0]).
static __device__ __forceinline__ unsigned lds_off(const void* p) {
  return (unsigned)(unsigned long long)p;
}

// GLOBAL_LOAD_ASYNC_TO_LDS_B128, GVS mode: mem = SGPR64 + VGPR32 (+0)
// Tracked by ASYNCcnt. vdst = LDS byte address, vaddr = byte offset, saddr = base.
static __device__ __forceinline__ void async_copy_b128(unsigned lds_addr,
                                                       unsigned long long base,
                                                       unsigned byte_off) {
  asm volatile("global_load_async_to_lds_b128 %0, %1, %2"
               :
               : "v"(lds_addr), "v"(byte_off), "s"(base)
               : "memory");
}

// ---------------------------------------------------------------------------
// f32 WMMA GEMM:  out[M x N] = alpha * (A[M x K] @ B[K x N])  (+ I if addId)
// Wave tile 64x64, block = 4 waves -> 128x128 block tile. All dims % 128 == 0.
// Uses V_WMMA_F32_16X16X4_F32 (full-precision matrix-power chain).
// ---------------------------------------------------------------------------
__global__ void __launch_bounds__(128)
gemm_f32_wmma_kernel(float* __restrict__ out, const float* __restrict__ A,
                     const float* __restrict__ B, int Mdim, int Ndim, int Kdim,
                     float alpha, int addIdentity) {
  (void)Mdim;
  const int lane = threadIdx.x & 31;
  const int wave = threadIdx.x >> 5;          // 0..3
  const int l15  = lane & 15;
  const int seg  = lane >> 4;                 // 0 or 1 (lane half)
  const int tileM = blockIdx.y * 128 + (wave >> 1) * 64;
  const int tileN = blockIdx.x * 128 + (wave & 1) * 64;

  const v8f zero = {0.f,0.f,0.f,0.f,0.f,0.f,0.f,0.f};
  v8f acc[4][4];
#pragma unroll
  for (int i = 0; i < 4; ++i)
#pragma unroll
    for (int j = 0; j < 4; ++j) acc[i][j] = zero;

  for (int k = 0; k < Kdim; k += 4) {
    v2f a[4], b[4];
#pragma unroll
    for (int mt = 0; mt < 4; ++mt) {
      // A frag: lane row = l15, K pair at 2*seg -> 8B aligned vector load (b64)
      a[mt] = *(const v2f*)(A + (size_t)(tileM + mt * 16 + l15) * Kdim + (k + 2 * seg));
    }
#pragma unroll
    for (int nt = 0; nt < 4; ++nt) {
      const float* bp = B + (size_t)(k + 2 * seg) * Ndim + (tileN + nt * 16 + l15);
      b[nt].x = bp[0];
      b[nt].y = bp[Ndim];
    }
#pragma unroll
    for (int mt = 0; mt < 4; ++mt)
#pragma unroll
      for (int nt = 0; nt < 4; ++nt)
        acc[mt][nt] = __builtin_amdgcn_wmma_f32_16x16x4_f32(
            false, a[mt], false, b[nt], (short)0, acc[mt][nt], false, false);
  }

#pragma unroll
  for (int mt = 0; mt < 4; ++mt) {
#pragma unroll
    for (int nt = 0; nt < 4; ++nt) {
      const int col = tileN + nt * 16 + l15;
#pragma unroll
      for (int j = 0; j < 8; ++j) {
        const int row = tileM + mt * 16 + seg * 8 + j;
        float v = alpha * acc[mt][nt][j];
        if (addIdentity && row == col) v += 1.0f;
        out[(size_t)row * Ndim + col] = v;
      }
    }
  }
}

// ---------------------------------------------------------------------------
// bf16 WMMA GEMM, LDS-staged with double-buffered async global->LDS copies.
//   out[M x N] = act[M x K] @ W^T (+ bias) (ReLU optional), W is [N x K] f32.
// Block = 4 waves = 128 threads, block tile 128x128, K-step 32.
// Per K-step each thread issues 16 global_load_async_to_lds_b128 (ASYNCcnt),
// overlapped with 16 V_WMMA_F32_16X16X32_BF16 per wave on the other buffer.
// f32 -> bf16 RNE conversion happens at LDS->VGPR fragment load time.
// Requires: M % 128 == 0, N % 128 == 0, K % 32 == 0.
// ---------------------------------------------------------------------------
template <bool RELU, bool HAS_BIAS>
__global__ void __launch_bounds__(128)
gemm_bf16_lds_kernel(float* __restrict__ out, const float* __restrict__ act,
                     const float* __restrict__ W, const float* __restrict__ bias,
                     int Mdim, int Ndim, int Kdim) {
  (void)Mdim;
  // Row pad 36 floats (144 B): keeps 16 B alignment for b128 LDS ops and
  // spreads the 64 banks across the 16 rows a fragment load touches.
  __shared__ float As[2][128][36];
  __shared__ float Bs[2][128][36];

  const int tid  = threadIdx.x;
  const int lane = tid & 31;
  const int wave = tid >> 5;
  const int l15  = lane & 15;
  const int seg  = lane >> 4;
  const int blockM = blockIdx.y * 128;
  const int blockN = blockIdx.x * 128;
  const int wm = (wave >> 1) * 64;   // wave tile origin inside block tile
  const int wn = (wave & 1) * 64;

  const unsigned long long actB = (unsigned long long)act;
  const unsigned long long wB   = (unsigned long long)W;
  const unsigned aRowBytes = (unsigned)((size_t)(blockM + tid) * Kdim * 4u);
  const unsigned bRowBytes = (unsigned)((size_t)(blockN + tid) * Kdim * 4u);

  const v8f zero = {0.f,0.f,0.f,0.f,0.f,0.f,0.f,0.f};
  v8f acc[4][4];
#pragma unroll
  for (int i = 0; i < 4; ++i)
#pragma unroll
    for (int j = 0; j < 4; ++j) acc[i][j] = zero;

  // Stage one 128x32 f32 tile of act and of W into LDS buffer `buf`.
  // thread t copies act row (blockM+t) and W row (blockN+t): 8 b128 each.
  auto stage = [&](int buf, int k0) {
    const unsigned la = lds_off(&As[buf][tid][0]);
    const unsigned lb = lds_off(&Bs[buf][tid][0]);
    const unsigned ga = aRowBytes + (unsigned)k0 * 4u;
    const unsigned gb = bRowBytes + (unsigned)k0 * 4u;
#pragma unroll
    for (int i = 0; i < 8; ++i) {
      async_copy_b128(la + 16u * i, actB, ga + 16u * i);
      async_copy_b128(lb + 16u * i, wB,   gb + 16u * i);
    }
  };

  const int Ksteps = Kdim / 32;
  stage(0, 0);

  for (int kt = 0; kt < Ksteps; ++kt) {
    const int buf = kt & 1;
    if (kt + 1 < Ksteps) {
      stage(buf ^ 1, (kt + 1) * 32);
      // 16 new asyncs in flight for buf^1; wait until only those remain
      // (async loads complete in order -> buffer `buf` is fully written).
      asm volatile("s_wait_asynccnt 0x10" ::: "memory");
    } else {
      asm volatile("s_wait_asynccnt 0x0" ::: "memory");
    }
    __syncthreads();   // make all waves' LDS writes visible

    v16bf a[4], b[4];
    // A fragments: lane covers K runs [8*seg, +8) and [16+8*seg, +8)
#pragma unroll
    for (int mt = 0; mt < 4; ++mt) {
      const float* row = &As[buf][wm + mt * 16 + l15][0];
      v16u bits;
#pragma unroll
      for (int e = 0; e < 8; ++e) bits[e]     = f32_to_bf16_rne(row[8 * seg + e]);
#pragma unroll
      for (int e = 0; e < 8; ++e) bits[8 + e] = f32_to_bf16_rne(row[16 + 8 * seg + e]);
      a[mt] = pack_bf16(bits);
    }
    // B fragments: lane covers K runs [8*seg, +8) and [16+8*seg, +8) at column l15
#pragma unroll
    for (int nt = 0; nt < 4; ++nt) {
      const float* row = &Bs[buf][wn + nt * 16 + l15][0];
      v16u bits;
#pragma unroll
      for (int e = 0; e < 8; ++e) bits[e]     = f32_to_bf16_rne(row[8 * seg + e]);
#pragma unroll
      for (int e = 0; e < 8; ++e) bits[8 + e] = f32_to_bf16_rne(row[16 + 8 * seg + e]);
      b[nt] = pack_bf16(bits);
    }

#pragma unroll
    for (int mt = 0; mt < 4; ++mt)
#pragma unroll
      for (int nt = 0; nt < 4; ++nt)
        acc[mt][nt] = __builtin_amdgcn_wmma_f32_16x16x32_bf16(
            false, a[mt], false, b[nt], (short)0, acc[mt][nt], false, false);

    __syncthreads();   // done reading `buf` before it is overwritten next iter
  }

#pragma unroll
  for (int mt = 0; mt < 4; ++mt) {
#pragma unroll
    for (int nt = 0; nt < 4; ++nt) {
      const int col = blockN + wn + nt * 16 + l15;
      const float bv = HAS_BIAS ? bias[col] : 0.0f;
#pragma unroll
      for (int j = 0; j < 8; ++j) {
        const int row = blockM + wm + mt * 16 + seg * 8 + j;
        float v = acc[mt][nt][j] + bv;
        if (RELU) v = v > 0.0f ? v : 0.0f;
        out[(size_t)row * Ndim + col] = v;
      }
    }
  }
}

// ---------------------------------------------------------------------------
// Square matrix transpose (n % 16 == 0), LDS-tiled. Block = 256 threads (16x16).
// ---------------------------------------------------------------------------
__global__ void __launch_bounds__(256)
transpose_kernel(float* __restrict__ out, const float* __restrict__ in, int n) {
  __shared__ float tile[16][17];
  const int bx = blockIdx.x * 16, by = blockIdx.y * 16;
  const int tx = threadIdx.x & 15, ty = threadIdx.x >> 4;
  tile[ty][tx] = in[(size_t)(by + ty) * n + (bx + tx)];
  __syncthreads();
  out[(size_t)(bx + ty) * n + (by + tx)] = tile[tx][ty];
}

// out = alpha*in + I   (n x n)
__global__ void __launch_bounds__(256)
scale_add_identity_kernel(float* __restrict__ out, const float* __restrict__ in,
                          float alpha, int n) {
  const int i = blockIdx.x * blockDim.x + threadIdx.x;
  if (i < n * n) {
    float v = alpha * in[i];
    if ((i / n) == (i % n)) v += 1.0f;
    out[i] = v;
  }
}

// ---------------------------------------------------------------------------
// Host-side orchestration.
//
// Math: RK4 on dz/dt = z@A^T is exactly z <- z @ T with
//   T = I + dt*N + dt^2/2 N^2 + dt^3/6 N^3 + dt^4/24 N^4,  N = A^T
// (Horner: T = I + dt*N*(I + dt/2*N*(I + dt/3*N*(I + dt/4*N))))
// 64 steps = T^64 via 6 squarings (f32 WMMA throughout). Then
//   y = MLP( X @ (P^T @ T^64) )  via bf16 WMMA with f32 accumulation.
// All batch GEMMs consume the B operand as [N x K] row-major, so C = P^T@T^64
// is transposed once (512x512, negligible) to reuse the single GEMM kernel.
// ---------------------------------------------------------------------------
extern "C" void kernel_launch(void* const* d_in, const int* in_sizes, int n_in,
                              void* d_out, int out_size, void* d_ws, size_t ws_size,
                              hipStream_t stream) {
  (void)in_sizes; (void)n_in; (void)out_size; (void)ws_size;

  const float* X  = (const float*)d_in[0];   // [B, Dx]
  const float* P  = (const float*)d_in[1];   // [Dz, Dx]
  const float* A  = (const float*)d_in[2];   // [Dz, Dz]
  const float* W1 = (const float*)d_in[3];   // [H, Dz]
  const float* b1 = (const float*)d_in[4];   // [H]
  const float* W2 = (const float*)d_in[5];   // [H, H]
  const float* b2 = (const float*)d_in[6];   // [H]
  const float* W3 = (const float*)d_in[7];   // [Dy, H]
  const float* b3 = (const float*)d_in[8];   // [Dy]
  float* y = (float*)d_out;                  // [B, Dy]

  const int B = 4096, Dx = 512, Dz = 512, Dy = 256, H = 2048;
  const float dt = 1.0f / 64.0f;             // (T1-T0)/n_steps, n_steps = 64

  // Workspace carve-up (f32): ~82 MB total
  float* w    = (float*)d_ws;
  float* Nm   = w;  w += (size_t)Dz * Dz;    // A^T
  float* t_a  = w;  w += (size_t)Dz * Dz;    // ping
  float* t_b  = w;  w += (size_t)Dz * Dz;    // pong
  float* Pt   = w;  w += (size_t)Dz * Dx;    // P^T
  float* Cm   = w;  w += (size_t)Dx * Dz;    // P^T @ T^64
  float* CmT  = w;  w += (size_t)Dz * Dx;    // (P^T @ T^64)^T   [N x K] form
  float* zT   = w;  w += (size_t)B * Dz;
  float* h1   = w;  w += (size_t)B * H;
  float* h2   = w;  w += (size_t)B * H;

  const dim3 tb(256);
  const dim3 tg(Dz / 16, Dz / 16);
  const dim3 gsmall(Dz / 128, Dz / 128);
  const int  n2 = Dz * Dz;

  // --- Build T = I + dt*N*(I + dt/2*N*(I + dt/3*N*(I + dt/4*N))) -----------
  transpose_kernel<<<tg, tb, 0, stream>>>(Nm, A, Dz);
  scale_add_identity_kernel<<<(n2 + 255) / 256, 256, 0, stream>>>(t_a, Nm, dt / 4.f, Dz);
  gemm_f32_wmma_kernel<<<gsmall, 128, 0, stream>>>(t_b, Nm, t_a, Dz, Dz, Dz, dt / 3.f, 1);
  gemm_f32_wmma_kernel<<<gsmall, 128, 0, stream>>>(t_a, Nm, t_b, Dz, Dz, Dz, dt / 2.f, 1);
  gemm_f32_wmma_kernel<<<gsmall, 128, 0, stream>>>(t_b, Nm, t_a, Dz, Dz, Dz, dt,       1);

  // --- T^64 via 6 squarings -------------------------------------------------
  float* src = t_b;
  float* dst = t_a;
  for (int s = 0; s < 6; ++s) {
    gemm_f32_wmma_kernel<<<gsmall, 128, 0, stream>>>(dst, src, src, Dz, Dz, Dz, 1.0f, 0);
    float* tmp = src; src = dst; dst = tmp;
  }
  // T^64 now in `src`

  // --- C = P^T @ T^64, then transpose to [N x K] weight form ---------------
  transpose_kernel<<<tg, tb, 0, stream>>>(Pt, P, Dz);
  gemm_f32_wmma_kernel<<<gsmall, 128, 0, stream>>>(Cm, Pt, src, Dx, Dz, Dz, 1.0f, 0);
  transpose_kernel<<<tg, tb, 0, stream>>>(CmT, Cm, Dz);

  // --- Batch side: zT = X @ CmT^T ; MLP head (bf16 WMMA + async LDS) -------
  gemm_bf16_lds_kernel<false, false>
      <<<dim3(Dz / 128, B / 128), 128, 0, stream>>>(zT, X, CmT, nullptr, B, Dz, Dx);
  gemm_bf16_lds_kernel<true, true>
      <<<dim3(H / 128, B / 128), 128, 0, stream>>>(h1, zT, W1, b1, B, H, Dz);
  gemm_bf16_lds_kernel<true, true>
      <<<dim3(H / 128, B / 128), 128, 0, stream>>>(h2, h1, W2, b2, B, H, H);
  gemm_bf16_lds_kernel<true, false>
      <<<dim3(Dy / 128, B / 128), 128, 0, stream>>>(y, h2, W3, b3, B, Dy, H);
}
